// SelfAttentionBlock_4973572129253
// MI455X (gfx1250) — matrix operand order repaired and checked by reading, not compiled, not run on hardware
//
#include <hip/hip_runtime.h>
#include <hip/hip_bf16.h>

// ---------------------------------------------------------------------------
// Self-attention block (B=4, H=W=64, C=256, d=32) for gfx1250 (MI455X).
// Flash-attention-style fused softmax-attention + bf16 WMMA GEMMs.
// V and weights are pre-transposed once so every LDS stage is b128-vectorized.
// Softmax row-sums computed on the WMMA pipe (P @ ones). ~36 MB of d_ws.
// ---------------------------------------------------------------------------

typedef __bf16 bf16;
typedef __attribute__((ext_vector_type(16))) __bf16 v16bf;
typedef __attribute__((ext_vector_type(8)))  __bf16 v8bf;
typedef __attribute__((ext_vector_type(8)))  float  v8f;
typedef __attribute__((ext_vector_type(4)))  unsigned int u32x4;

#define BB 4
#define NN 4096
#define CC 256
#define DD 32

__device__ __forceinline__ unsigned short f2bs(float f) {
  union { float f; unsigned u; } a; a.f = f;
  unsigned r = a.u + 0x7FFFu + ((a.u >> 16) & 1u);   // round-to-nearest-even
  return (unsigned short)(r >> 16);
}
__device__ __forceinline__ bf16 f2bf(float f) {
  union { unsigned short s; bf16 b; } o; o.s = f2bs(f);
  return o.b;
}
__device__ __forceinline__ v8f vzero8() {
  v8f z;
#pragma unroll
  for (int i = 0; i < 8; ++i) z[i] = 0.0f;
  return z;
}
__device__ __forceinline__ u32x4 uzero4() {
  u32x4 z;
#pragma unroll
  for (int i = 0; i < 4; ++i) z[i] = 0u;
  return z;
}

// ---- WMMA fragment loaders (ISA 7.12.2 layouts, wave32) --------------------
__device__ __forceinline__ v16bf load_a_frag(const bf16* p, int row, int rowLen,
                                             int kbase, int hf) {
  const v8bf* p0 = (const v8bf*)(p + row * rowLen + kbase + hf * 8);
  const v8bf* p1 = (const v8bf*)(p + row * rowLen + kbase + hf * 8 + 16);
  v8bf lo = *p0, hi = *p1;
  v16bf r;
#pragma unroll
  for (int e = 0; e < 8; ++e) { r[e] = lo[e]; r[8 + e] = hi[e]; }
  return r;
}
__device__ __forceinline__ v16bf load_b_frag(const bf16* pT, int n, int rowLen,
                                             int kbase, int hf) {
  const v8bf* p0 = (const v8bf*)(pT + n * rowLen + kbase + hf * 8);
  const v8bf* p1 = (const v8bf*)(pT + n * rowLen + kbase + hf * 8 + 16);
  v8bf lo = *p0, hi = *p1;
  v16bf r;
#pragma unroll
  for (int e = 0; e < 8; ++e) { r[e] = lo[e]; r[8 + e] = hi[e]; }
  return r;
}

// ---- fp32 -> bf16 conversion (row-major) -----------------------------------
__global__ void convert_f32_bf16(const float* __restrict__ in,
                                 bf16* __restrict__ out, int n) {
  int i = blockIdx.x * blockDim.x + threadIdx.x;
  int stride = gridDim.x * blockDim.x;
  for (; i < n; i += stride) out[i] = f2bf(in[i]);
}

// ---- fp32 W[K x Ncol] -> bf16 W^T[Ncol x K] --------------------------------
__global__ void convert_transpose(const float* __restrict__ W,
                                  bf16* __restrict__ WT, int K, int Ncol) {
  int i = blockIdx.x * blockDim.x + threadIdx.x;
  if (i < K * Ncol) {
    int n = i / K, k = i - n * K;
    WT[i] = f2bf(W[(size_t)k * Ncol + n]);
  }
}

// ---- bf16 v[B*N x C] -> vT[B x C x N] (64x64 LDS tile transpose) -----------
__global__ __launch_bounds__(256) void transpose_v(const bf16* __restrict__ v,
                                                   bf16* __restrict__ vT) {
  __shared__ unsigned short s[64][65];  // +1 pad: conflict-free column reads
  const int b = blockIdx.z;
  const int jb = blockIdx.x * 64, cb = blockIdx.y * 64;
  const int tid = threadIdx.x;
  {
    int r = tid >> 2, c0 = (tid & 3) * 16;
    const u32x4* g =
        (const u32x4*)(v + ((size_t)(b * NN + jb + r)) * CC + cb + c0);
    union { u32x4 q; unsigned short h[8]; } a0, a1;
    a0.q = g[0]; a1.q = g[1];
#pragma unroll
    for (int e = 0; e < 8; ++e) {
      s[r][c0 + e] = a0.h[e];
      s[r][c0 + 8 + e] = a1.h[e];
    }
  }
  __syncthreads();
  {
    int cr = tid >> 2, j0 = (tid & 3) * 16;
    union { u32x4 q; unsigned short h[8]; } o0, o1;
#pragma unroll
    for (int e = 0; e < 8; ++e) {
      o0.h[e] = s[j0 + e][cr];
      o1.h[e] = s[j0 + 8 + e][cr];
    }
    u32x4* d = (u32x4*)(vT + ((size_t)(b * CC + cb + cr)) * NN + jb + j0);
    d[0] = o0.q;
    d[1] = o1.q;
  }
}

// ---- bf16 WMMA GEMM: out[M x Ncol] = A[M x K] @ W + bias -------------------
// Weights given pre-transposed: BT[Ncol x K].
template <bool FINAL>
__global__ __launch_bounds__(128) void gemm_bf16(
    const bf16* __restrict__ A, const bf16* __restrict__ BT,
    const float* __restrict__ bias, bf16* __restrict__ outb,
    float* __restrict__ outf, const float* __restrict__ resid,
    const float* __restrict__ gamma, int M, int K, int Ncol) {
  __shared__ unsigned short sA[64 * 32];
  __shared__ unsigned short sB[64 * 32];  // transposed: [n][k]
  const int tid = threadIdx.x;
  const int wv = tid >> 5, lane = tid & 31, hf = lane >> 4, ln = lane & 15;
  const int rowbase = blockIdx.x * 64;
  const int colbase = blockIdx.y * 64;
  const int ntc = (Ncol - colbase) >= 64 ? 4 : ((Ncol - colbase) >> 4);

  v8f acc[4];
#pragma unroll
  for (int t = 0; t < 4; ++t) acc[t] = vzero8();

  for (int kb = 0; kb < K; kb += 32) {
    __syncthreads();
    {  // A tile 64x32: 16 bf16 per thread, 2 x b128
      int row = tid >> 1, cc = (tid & 1) * 16;
      const u32x4* g = (const u32x4*)(A + (size_t)(rowbase + row) * K + kb + cc);
      *(u32x4*)&sA[row * 32 + cc] = g[0];
      *(u32x4*)&sA[row * 32 + cc + 8] = g[1];
    }
    {  // B tile (pre-transposed in global): coalesced 2 x b128
      int nl = tid >> 1, cc = (tid & 1) * 16;
      int col = colbase + nl;
      u32x4 d0 = uzero4(), d1 = uzero4();
      if (col < Ncol) {
        const u32x4* g = (const u32x4*)(BT + (size_t)col * K + kb + cc);
        d0 = g[0];
        d1 = g[1];
      }
      *(u32x4*)&sB[nl * 32 + cc] = d0;
      *(u32x4*)&sB[nl * 32 + cc + 8] = d1;
    }
    __syncthreads();
    v16bf af = load_a_frag((const bf16*)sA, wv * 16 + ln, 32, 0, hf);
    {  // t = 0 (no reuse hint on the first of the A-sharing chain)
      v16bf bf_ = load_b_frag((const bf16*)sB, ln, 32, 0, hf);
      acc[0] = __builtin_amdgcn_wmma_f32_16x16x32_bf16(
          false, af, false, bf_, (short)0, acc[0], false, false);
    }
#pragma unroll
    for (int t = 1; t < 4; ++t) {
      if (t < ntc) {
        v16bf bf_ = load_b_frag((const bf16*)sB, t * 16 + ln, 32, 0, hf);
        acc[t] = __builtin_amdgcn_wmma_f32_16x16x32_bf16(
            false, af, false, bf_, (short)0, acc[t], true, false);
      }
    }
  }

#pragma unroll
  for (int t = 0; t < 4; ++t) {
    if (t >= ntc) continue;
    int col = colbase + t * 16 + ln;
    float bsc = bias[col];
#pragma unroll
    for (int v = 0; v < 8; ++v) {
      size_t row = (size_t)rowbase + wv * 16 + v + 8 * hf;
      float val = acc[t][v] + bsc;
      if (FINAL) {
        outf[row * Ncol + col] = gamma[0] * val + resid[row * Ncol + col];
      } else {
        outb[row * Ncol + col] = f2bf(val);
      }
    }
  }
}

// ---- fused flash attention -------------------------------------------------
// rel[b,i,j] = k_i . q_j ; softmax over j ; o[b,i,:] = sum_j w * v[b,j,:]
// Block = 256 threads (8 waves) handles 128 output rows i; streams 64-wide
// j tiles. V comes pre-transposed: vT[b][c][j].
__global__ __launch_bounds__(256) void attn_kernel(
    const bf16* __restrict__ qg, const bf16* __restrict__ kg,
    const bf16* __restrict__ vTg, bf16* __restrict__ og) {
  __shared__ unsigned short sK[128 * 32];       // k rows (the "queries")
  __shared__ unsigned short sQ[64 * 32];        // q rows (B-layout for logits)
  __shared__ unsigned short sVT[256 * 64];      // V^T tile [c][j]
  __shared__ unsigned short sP[8][16 * 64];     // per-wave P restage

  const int tid = threadIdx.x;
  const int wv = tid >> 5, lane = tid & 31, hf = lane >> 4, ln = lane & 15;
  const int b = blockIdx.x >> 5;  // 32 row-tiles of 128 per batch
  const int it = blockIdx.x & 31;
  const size_t kqbase = (size_t)b * NN * DD;
  const int ibase = it * 128;

  {  // K tile (128 rows) -> row-major LDS, 2 x b128 per thread
    int row = tid >> 1, cc = (tid & 1) * 16;
    const u32x4* g =
        (const u32x4*)(kg + kqbase + (size_t)(ibase + row) * DD + cc);
    *(u32x4*)&sK[row * 32 + cc] = g[0];
    *(u32x4*)&sK[row * 32 + cc + 8] = g[1];
  }
  __syncthreads();
  const v16bf afS = load_a_frag((const bf16*)sK, wv * 16 + ln, 32, 0, hf);

  // all-ones B fragment: row-sum of P via WMMA (D[m][n] = sum_k P[m][k])
  v16bf ones;
  {
    union { unsigned short s; bf16 b; } ob_; ob_.s = 0x3F80;  // 1.0 bf16
#pragma unroll
    for (int e = 0; e < 16; ++e) ones[e] = ob_.b;
  }

  float mrun[8], lrun[8];
  v8f oacc[16];
#pragma unroll
  for (int v = 0; v < 8; ++v) { mrun[v] = -3e38f; lrun[v] = 0.0f; }
#pragma unroll
  for (int ct = 0; ct < 16; ++ct) oacc[ct] = vzero8();

  for (int jt = 0; jt < NN / 64; ++jt) {
    const int jbase = jt * 64;
    __syncthreads();
    {  // Q tile rows jbase..+63 -> sQ (row-major == transposed-B layout)
      int row = tid >> 2, seg = (tid & 3) * 8;
      const u32x4* g =
          (const u32x4*)(qg + kqbase + (size_t)(jbase + row) * DD + seg);
      *(u32x4*)&sQ[row * 32 + seg] = g[0];
    }
    {  // V^T tile: row c, 64 contiguous j -> pure b128 copies
      int c = tid;
      const bf16* vrow = vTg + ((size_t)(b * CC + c)) * NN + jbase;
      const u32x4* g = (const u32x4*)vrow;
      if (jt + 1 < NN / 64)  // prefetch next tile (global_prefetch_b8)
        __builtin_prefetch((const void*)(vrow + 64), 0, 1);
#pragma unroll
      for (int u = 0; u < 8; ++u) *(u32x4*)&sVT[c * 64 + u * 8] = g[u];
    }
    __syncthreads();

    // logits S[i, jbase + t*16 + n] = k_i . q_j (shared A -> reuse_a hint)
    v8f s[4];
    {
      v16bf bq = load_b_frag((const bf16*)sQ, ln, 32, 0, hf);
      s[0] = __builtin_amdgcn_wmma_f32_16x16x32_bf16(
          false, afS, false, bq, (short)0, vzero8(), false, false);
    }
#pragma unroll
    for (int t = 1; t < 4; ++t) {
      v16bf bq = load_b_frag((const bf16*)sQ, t * 16 + ln, 32, 0, hf);
      s[t] = __builtin_amdgcn_wmma_f32_16x16x32_bf16(
          false, afS, false, bq, (short)0, vzero8(), true, false);
    }

    // ---- row max: per-lane partial, then 4 batched xor-shuffle stages ----
    float pm[8];
#pragma unroll
    for (int v = 0; v < 8; ++v)
      pm[v] = fmaxf(fmaxf(s[0][v], s[1][v]), fmaxf(s[2][v], s[3][v]));
#pragma unroll
    for (int m = 1; m < 16; m <<= 1) {
      float t0[8];
#pragma unroll
      for (int v = 0; v < 8; ++v) t0[v] = __shfl_xor(pm[v], m, 32);
#pragma unroll
      for (int v = 0; v < 8; ++v) pm[v] = fmaxf(pm[v], t0[v]);
    }

    // ---- rescale only when a row max actually grows (uniform branch) ----
    unsigned grew = 0;
#pragma unroll
    for (int v = 0; v < 8; ++v) grew |= (pm[v] > mrun[v]) ? 1u : 0u;
    if (__ballot(grew) != 0ull) {
#pragma unroll
      for (int v = 0; v < 8; ++v) {
        float mnew = fmaxf(mrun[v], pm[v]);
        float alpha = __expf(mrun[v] - mnew);
        mrun[v] = mnew;
        lrun[v] *= alpha;
#pragma unroll
        for (int ct = 0; ct < 16; ++ct) oacc[ct][v] *= alpha;
      }
    }

    // ---- P = exp(S - m), restage to LDS in A layout ----
#pragma unroll
    for (int v = 0; v < 8; ++v) {
      int prow = v + 8 * hf;
#pragma unroll
      for (int t = 0; t < 4; ++t)
        sP[wv][prow * 64 + t * 16 + ln] = f2bs(__expf(s[t][v] - mrun[v]));
    }

    // ---- O += P @ V and row-sum(P) on the WMMA pipe ----
    v8f rsum = vzero8();
#pragma unroll
    for (int ss = 0; ss < 2; ++ss) {
      v16bf pf = load_a_frag((const bf16*)sP[wv], ln, 64, ss * 32, hf);
      rsum = __builtin_amdgcn_wmma_f32_16x16x32_bf16(
          false, pf, false, ones, (short)0, rsum, false, false);
#pragma unroll
      for (int ct = 0; ct < 16; ++ct) {
        v16bf vf = load_b_frag((const bf16*)sVT, ct * 16 + ln, 64, ss * 32, hf);
        oacc[ct] = __builtin_amdgcn_wmma_f32_16x16x32_bf16(
            false, pf, false, vf, (short)0, oacc[ct], true, false);
      }
    }
#pragma unroll
    for (int v = 0; v < 8; ++v) lrun[v] += rsum[v];
  }

  // normalize and write o (bf16)
#pragma unroll
  for (int v = 0; v < 8; ++v) {
    float inv = 1.0f / lrun[v];
    size_t row = (size_t)b * NN + ibase + wv * 16 + v + 8 * hf;
#pragma unroll
    for (int ct = 0; ct < 16; ++ct)
      og[row * CC + ct * 16 + ln] = f2bf(oacc[ct][v] * inv);
  }
}

// ---------------------------------------------------------------------------
extern "C" void kernel_launch(void* const* d_in, const int* in_sizes, int n_in,
                              void* d_out, int out_size, void* d_ws,
                              size_t ws_size, hipStream_t stream) {
  const float* x     = (const float*)d_in[0];
  const float* Wq    = (const float*)d_in[1];
  const float* bq    = (const float*)d_in[2];
  const float* Wk    = (const float*)d_in[3];
  const float* bk    = (const float*)d_in[4];
  const float* Wv    = (const float*)d_in[5];
  const float* bv    = (const float*)d_in[6];
  const float* Wo    = (const float*)d_in[7];
  const float* bo    = (const float*)d_in[8];
  const float* gamma = (const float*)d_in[9];
  float* out = (float*)d_out;

  char* ws = (char*)d_ws;
  size_t off = 0;
  auto alloc = [&](size_t bytes) {
    void* p = ws + off;
    off = (off + bytes + 255) & ~(size_t)255;
    return p;
  };
  bf16* xb   = (bf16*)alloc((size_t)BB * NN * CC * 2);
  bf16* wqT  = (bf16*)alloc((size_t)CC * DD * 2);   // [DD x CC] = Wq^T
  bf16* wkT  = (bf16*)alloc((size_t)CC * DD * 2);
  bf16* wvT  = (bf16*)alloc((size_t)CC * CC * 2);
  bf16* woT  = (bf16*)alloc((size_t)CC * CC * 2);
  bf16* qb   = (bf16*)alloc((size_t)BB * NN * DD * 2);
  bf16* kb   = (bf16*)alloc((size_t)BB * NN * DD * 2);
  bf16* vb   = (bf16*)alloc((size_t)BB * NN * CC * 2);
  bf16* vTb  = (bf16*)alloc((size_t)BB * NN * CC * 2);
  bf16* ob   = (bf16*)alloc((size_t)BB * NN * CC * 2);

  auto cgrid = [](int n) { return dim3((unsigned)((n + 255) / 256)); };
  convert_f32_bf16<<<cgrid(BB * NN * CC), 256, 0, stream>>>(x, xb,
                                                            BB * NN * CC);
  convert_transpose<<<cgrid(CC * DD), 256, 0, stream>>>(Wq, wqT, CC, DD);
  convert_transpose<<<cgrid(CC * DD), 256, 0, stream>>>(Wk, wkT, CC, DD);
  convert_transpose<<<cgrid(CC * CC), 256, 0, stream>>>(Wv, wvT, CC, CC);
  convert_transpose<<<cgrid(CC * CC), 256, 0, stream>>>(Wo, woT, CC, CC);

  const int M = BB * NN;  // 16384
  dim3 blk(128);
  gemm_bf16<false><<<dim3(M / 64, 1), blk, 0, stream>>>(
      xb, wqT, bq, qb, nullptr, nullptr, nullptr, M, CC, DD);
  gemm_bf16<false><<<dim3(M / 64, 1), blk, 0, stream>>>(
      xb, wkT, bk, kb, nullptr, nullptr, nullptr, M, CC, DD);
  gemm_bf16<false><<<dim3(M / 64, 4), blk, 0, stream>>>(
      xb, wvT, bv, vb, nullptr, nullptr, nullptr, M, CC, CC);

  transpose_v<<<dim3(NN / 64, CC / 64, BB), 256, 0, stream>>>(vb, vTb);

  attn_kernel<<<dim3(BB * (NN / 128)), 256, 0, stream>>>(qb, kb, vTb, ob);

  gemm_bf16<true><<<dim3(M / 64, 4), blk, 0, stream>>>(
      ob, woT, bo, nullptr, out, x, gamma, M, CC, CC);
}
